// AutoregressivePredictor_48919677502034
// MI455X (gfx1250) — matrix operand order repaired and checked by reading, not compiled, not run on hardware
//
#include <hip/hip_runtime.h>
#include <hip/hip_bf16.h>
#include <math.h>

// ---------------------------------------------------------------------------
// Llama-3-8B single-layer greedy 8-step decode, KV-cached (exact under the
// causal mask), bf16 WMMA (v_wmma_f32_16x16x32_bf16), M=32 batched GEMMs.
// MI455X: weight streaming (~440MB/step, > 192MB L2) bounds runtime, so GEMMs
// keep weights bf16, read them once per step with coalesced b128 loads, and
// hide fill latency with double-buffered LDS + TDM (tensor_load_to_lds) for
// the activation tile.
// ---------------------------------------------------------------------------

typedef __bf16 bf16;
typedef __attribute__((ext_vector_type(16))) __bf16 v16bf;
typedef __attribute__((ext_vector_type(8)))  __bf16 v8bf;
typedef __attribute__((ext_vector_type(4)))  __bf16 v4bf;
typedef __attribute__((ext_vector_type(8)))  float  v8f;
typedef __attribute__((ext_vector_type(4)))  unsigned int v4u;
typedef __attribute__((ext_vector_type(8)))  int    v8i_;
typedef __attribute__((ext_vector_type(4)))  int    v4i_;

#define DMODEL 4096
#define NHEADS 32
#define NKV    8
#define HDIM   128
#define DFF    14336
#define VOCAB  1024
#define NSTEPS 8
#define ROWS   32
#define KVSTRIDE (NSTEPS * NKV * HDIM)

#if defined(__has_builtin)
#if __has_builtin(__builtin_amdgcn_tensor_load_to_lds) && __has_builtin(__builtin_amdgcn_s_wait_tensorcnt)
#define HAVE_TDM 1
#endif
#endif
#ifndef HAVE_TDM
#define HAVE_TDM 0
#endif

// Stage W[k0..k0+63][bn..bn+127] transposed into lds[n][k] (stride 72).
// Each thread: 4 consecutive K-rows x 8 N-cols -> 8 packed ds_store_b64.
__device__ __forceinline__ void stage_w(const bf16* __restrict__ W, int N,
                                        int k0, int bn, int tid,
                                        bf16 (*lds)[72]) {
  const int kr4  = (tid >> 4) * 4;
  const int colb = (tid & 15) * 8;
  const bf16* wp = W + (size_t)(k0 + kr4) * N + bn + colb;
  v8bf w0 = *(const v8bf*)(wp);
  v8bf w1 = *(const v8bf*)(wp + (size_t)N);
  v8bf w2 = *(const v8bf*)(wp + (size_t)2 * N);
  v8bf w3 = *(const v8bf*)(wp + (size_t)3 * N);
  #pragma unroll
  for (int j = 0; j < 8; ++j) {
    v4bf p = { w0[j], w1[j], w2[j], w3[j] };
    *(v4bf*)&lds[colb + j][kr4] = p;
  }
}

#if HAVE_TDM
// TDM: DMA A[0..31][k0..k0+63] (row-major, row stride K elements) into LDS at
// lds_off, padding +4 DWORDs after every 32 DWORDs -> LDS row stride 72 bf16.
__device__ __forceinline__ void tdm_load_A(const bf16* __restrict__ A, int K,
                                           int k0, unsigned int lds_off) {
  unsigned long long ga = (unsigned long long)(uintptr_t)A + (unsigned long long)k0 * 2ull;
  v4u g0 = { 1u,                                  // count=1, user desc
             lds_off,                             // LDS byte address
             (unsigned int)ga,                    // global_addr[31:0]
             ((unsigned int)(ga >> 32) & 0x01ffffffu) | (2u << 30) };  // [56:32] | type=2
  v8i_ g1 = { (int)((1u << 16)                    // data_size = 2B
                  | (1u << 20)                    // pad_enable
                  | (4u << 22)                    // pad_interval: 32 DWORDs
                  | (3u << 25)),                  // pad_amount: 4 DWORDs
              (int)(((unsigned int)K & 0xffffu) << 16),          // tensor_dim0 lo
              (int)(((unsigned int)K >> 16) | (32u << 16)),      // dim0 hi | tensor_dim1=32
              (int)(64u << 16),                                  // tile_dim0 = 64
              32,                                                // tile_dim1 = 32
              K,                                                 // tensor_dim0_stride lo
              0, 0 };
  v4i_ z4 = { 0, 0, 0, 0 };
  v8i_ z8 = { 0, 0, 0, 0, 0, 0, 0, 0 };
  __builtin_amdgcn_tensor_load_to_lds(g0, g1, z4, z4, z8, 0);
}
#endif

// ---------------------------------------------------------------------------
// Generic GEMM: C[32 x N] = A[32 x K] * W[K x N]  (bf16 row-major)
// mode 0: bf16 out   mode 1: bf16 out + residual   mode 2: f32 out + bias
// Block = 8 waves = 32x128 tile; double-buffered LDS; TDM for A when present.
// ---------------------------------------------------------------------------
__global__ __launch_bounds__(256) void gemm32(
    const bf16* __restrict__ A, const bf16* __restrict__ W,
    int K, int N,
    bf16* __restrict__ Cb, float* __restrict__ Cf,
    const bf16* __restrict__ resid, const bf16* __restrict__ bias,
    int ldc, int mode)
{
  __shared__ __align__(16) bf16 ldsW[2][128][72];
#if HAVE_TDM
  __shared__ __align__(16) bf16 ldsA[2][32][72];
#endif
  const int tid  = threadIdx.x;
  const int lane = tid & 31;
  const int wv   = tid >> 5;
  const int bn   = blockIdx.x * 128;
  const int half = lane >> 4;
  const int mrow = lane & 15;
  const int NC   = K / 64;

  v8f acc0 = {}; v8f acc1 = {};

  stage_w(W, N, 0, bn, tid, ldsW[0]);
#if HAVE_TDM
  if (wv == 0) tdm_load_A(A, K, 0, (unsigned int)(uintptr_t)&ldsA[0][0][0]);
#endif

  for (int c = 0; c < NC; ++c) {
    const int cur = c & 1, nxt = cur ^ 1;
#if HAVE_TDM
    if (wv == 0) __builtin_amdgcn_s_wait_tensorcnt(0);
#endif
    __syncthreads();                          // chunk c staged; prior reads done
    if (c + 1 < NC) {
      stage_w(W, N, (c + 1) * 64, bn, tid, ldsW[nxt]);
#if HAVE_TDM
      if (wv == 0) tdm_load_A(A, K, (c + 1) * 64, (unsigned int)(uintptr_t)&ldsA[nxt][0][0]);
#endif
    }
    const int k0 = c * 64;
    #pragma unroll
    for (int kk = 0; kk < 64; kk += 32) {
      v16bf bfrag;
      const bf16* bp = &ldsW[cur][wv * 16 + mrow][kk + half * 8];
      ((v8bf*)&bfrag)[0] = *(const v8bf*)(bp);
      ((v8bf*)&bfrag)[1] = *(const v8bf*)(bp + 16);
      #pragma unroll
      for (int mt = 0; mt < 2; ++mt) {
        v16bf afrag;
#if HAVE_TDM
        const bf16* ap = &ldsA[cur][mt * 16 + mrow][kk + half * 8];
#else
        const bf16* ap = A + (size_t)(mt * 16 + mrow) * K + k0 + kk + half * 8;
#endif
        ((v8bf*)&afrag)[0] = *(const v8bf*)(ap);
        ((v8bf*)&afrag)[1] = *(const v8bf*)(ap + 16);
        if (mt == 0)
          acc0 = __builtin_amdgcn_wmma_f32_16x16x32_bf16(
                     false, afrag, false, bfrag, (short)0, acc0, false, false);
        else
          acc1 = __builtin_amdgcn_wmma_f32_16x16x32_bf16(
                     false, afrag, false, bfrag, (short)0, acc1, false, false);
      }
    }
  }

  const int ncol = bn + wv * 16 + mrow;
  #pragma unroll
  for (int mt = 0; mt < 2; ++mt) {
    v8f acc = mt ? acc1 : acc0;
    #pragma unroll
    for (int r = 0; r < 8; ++r) {
      int m = mt * 16 + half * 8 + r;
      float v = acc[r];
      if (mode == 2) {
        Cf[(size_t)m * ldc + ncol] = v + (float)bias[ncol];
      } else {
        if (mode == 1) v += (float)resid[(size_t)m * ldc + ncol];
        Cb[(size_t)m * ldc + ncol] = (bf16)v;
      }
    }
  }
}

// ---------------------------------------------------------------------------
// Fused gate/up: GU[32 x DFF] = silu(A*Wg) * (A*Wu); shares A fragments.
// Single-buffered W tiles (two of them), TDM double-buffered A.
// ---------------------------------------------------------------------------
__global__ __launch_bounds__(256) void gemm_gateup(
    const bf16* __restrict__ A, const bf16* __restrict__ Wg,
    const bf16* __restrict__ Wu, bf16* __restrict__ GU)
{
  __shared__ __align__(16) bf16 ldsG[128][72];
  __shared__ __align__(16) bf16 ldsU[128][72];
#if HAVE_TDM
  __shared__ __align__(16) bf16 ldsA[2][32][72];
#endif
  const int tid  = threadIdx.x;
  const int lane = tid & 31;
  const int wv   = tid >> 5;
  const int bn   = blockIdx.x * 128;
  const int half = lane >> 4;
  const int mrow = lane & 15;

  v8f g0 = {}, g1 = {}, u0 = {}, u1 = {};

#if HAVE_TDM
  if (wv == 0) tdm_load_A(A, DMODEL, 0, (unsigned int)(uintptr_t)&ldsA[0][0][0]);
#endif

  for (int c = 0; c < DMODEL / 64; ++c) {
    const int k0 = c * 64;
    stage_w(Wg, DFF, k0, bn, tid, ldsG);
    stage_w(Wu, DFF, k0, bn, tid, ldsU);
#if HAVE_TDM
    const int cur = c & 1, nxt = cur ^ 1;
    if (wv == 0) {
      __builtin_amdgcn_s_wait_tensorcnt(0);
      if (c + 1 < DMODEL / 64)
        tdm_load_A(A, DMODEL, (c + 1) * 64, (unsigned int)(uintptr_t)&ldsA[nxt][0][0]);
    }
#endif
    __syncthreads();
    #pragma unroll
    for (int kk = 0; kk < 64; kk += 32) {
      v16bf bg, bu;
      const bf16* gp = &ldsG[wv * 16 + mrow][kk + half * 8];
      const bf16* up = &ldsU[wv * 16 + mrow][kk + half * 8];
      ((v8bf*)&bg)[0] = *(const v8bf*)(gp);  ((v8bf*)&bg)[1] = *(const v8bf*)(gp + 16);
      ((v8bf*)&bu)[0] = *(const v8bf*)(up);  ((v8bf*)&bu)[1] = *(const v8bf*)(up + 16);
      #pragma unroll
      for (int mt = 0; mt < 2; ++mt) {
        v16bf af;
#if HAVE_TDM
        const bf16* ap = &ldsA[cur][mt * 16 + mrow][kk + half * 8];
#else
        const bf16* ap = A + (size_t)(mt * 16 + mrow) * DMODEL + k0 + kk + half * 8;
#endif
        ((v8bf*)&af)[0] = *(const v8bf*)(ap);
        ((v8bf*)&af)[1] = *(const v8bf*)(ap + 16);
        if (mt == 0) {
          g0 = __builtin_amdgcn_wmma_f32_16x16x32_bf16(false, af, false, bg, (short)0, g0, false, false);
          u0 = __builtin_amdgcn_wmma_f32_16x16x32_bf16(false, af, false, bu, (short)0, u0, false, false);
        } else {
          g1 = __builtin_amdgcn_wmma_f32_16x16x32_bf16(false, af, false, bg, (short)0, g1, false, false);
          u1 = __builtin_amdgcn_wmma_f32_16x16x32_bf16(false, af, false, bu, (short)0, u1, false, false);
        }
      }
    }
    __syncthreads();
  }

  const int ncol = bn + wv * 16 + mrow;
  #pragma unroll
  for (int mt = 0; mt < 2; ++mt) {
    v8f ga = mt ? g1 : g0;
    v8f ua = mt ? u1 : u0;
    #pragma unroll
    for (int r = 0; r < 8; ++r) {
      int m = mt * 16 + half * 8 + r;
      float g = ga[r], u = ua[r];
      float s = g / (1.0f + __expf(-g));            // silu
      GU[(size_t)m * DFF + ncol] = (bf16)(s * u);
    }
  }
}

// ---------------------------------------------------------------------------
__global__ __launch_bounds__(256) void rmsnorm_kernel(
    const bf16* __restrict__ x, const bf16* __restrict__ w, bf16* __restrict__ out)
{
  const int r = blockIdx.x, t = threadIdx.x;
  __shared__ float red[256];
  float ss = 0.f;
  for (int j = t; j < DMODEL; j += 256) { float v = (float)x[(size_t)r * DMODEL + j]; ss += v * v; }
  red[t] = ss; __syncthreads();
  for (int off = 128; off > 0; off >>= 1) { if (t < off) red[t] += red[t + off]; __syncthreads(); }
  const float inv = rsqrtf(red[0] / (float)DMODEL + 1e-5f);
  for (int j = t; j < DMODEL; j += 256) {
    float xn = (float)x[(size_t)r * DMODEL + j] * inv;
    bf16 xb = (bf16)xn;
    out[(size_t)r * DMODEL + j] = (bf16)((float)w[j] * (float)xb);
  }
}

__global__ __launch_bounds__(256) void rope_kernel(
    const bf16* __restrict__ qr, const bf16* __restrict__ kr,
    bf16* __restrict__ q, bf16* __restrict__ kc, int pos)
{
  const int r = blockIdx.x, t = threadIdx.x;
  const float LT64 = 13.122363377404328f / 64.0f;   // ln(5e5)/64
  for (int j = t; j < NHEADS * HDIM; j += 256) {
    int hd = j & (HDIM - 1);
    int i  = hd & 63;
    float ang = (float)pos * __expf(-(float)i * LT64);
    float c = __cosf(ang), s = __sinf(ang);
    float x  = (float)qr[(size_t)r * DMODEL + j];
    float xp = (float)qr[(size_t)r * DMODEL + (hd < 64 ? j + 64 : j - 64)];
    q[(size_t)r * DMODEL + j] = (bf16)((hd < 64) ? (x * c - xp * s) : (x * c + xp * s));
  }
  for (int j = t; j < NKV * HDIM; j += 256) {
    int hd = j & (HDIM - 1);
    int i  = hd & 63;
    float ang = (float)pos * __expf(-(float)i * LT64);
    float c = __cosf(ang), s = __sinf(ang);
    float x  = (float)kr[(size_t)r * NKV * HDIM + j];
    float xp = (float)kr[(size_t)r * NKV * HDIM + (hd < 64 ? j + 64 : j - 64)];
    kc[(size_t)r * KVSTRIDE + (size_t)pos * NKV * HDIM + j] =
        (bf16)((hd < 64) ? (x * c - xp * s) : (x * c + xp * s));
  }
}

__global__ __launch_bounds__(128) void attn_kernel(
    const bf16* __restrict__ q, const bf16* __restrict__ kc,
    const bf16* __restrict__ vc, bf16* __restrict__ ctx, int s)
{
  const int r = blockIdx.x, h = blockIdx.y, t = threadIdx.x;
  const int kvh = h >> 2;
  __shared__ float red[128];
  __shared__ float scores[NSTEPS];
  const float qv = (float)q[(size_t)r * DMODEL + h * HDIM + t];

  for (int i = 0; i <= s; ++i) {
    float kv = (float)kc[(size_t)r * KVSTRIDE + (size_t)i * NKV * HDIM + kvh * HDIM + t];
    red[t] = qv * kv; __syncthreads();
    for (int off = 64; off > 0; off >>= 1) { if (t < off) red[t] += red[t + off]; __syncthreads(); }
    if (t == 0) scores[i] = red[0] * 0.08838834764831845f;   // 1/sqrt(128)
    __syncthreads();
  }
  float mx = -1e30f;
  for (int i = 0; i <= s; ++i) mx = fmaxf(mx, scores[i]);
  float p[NSTEPS]; float den = 0.f;
  for (int i = 0; i <= s; ++i) { p[i] = __expf(scores[i] - mx); den += p[i]; }
  const float inv = 1.0f / den;
  float acc = 0.f;
  for (int i = 0; i <= s; ++i) {
    bf16 pb = (bf16)(p[i] * inv);
    float vv = (float)vc[(size_t)r * KVSTRIDE + (size_t)i * NKV * HDIM + kvh * HDIM + t];
    acc += (float)pb * vv;
  }
  ctx[(size_t)r * DMODEL + h * HDIM + t] = (bf16)acc;
}

__global__ __launch_bounds__(256) void argmax_embed_kernel(
    const float* __restrict__ logits, const bf16* __restrict__ emb,
    bf16* __restrict__ h, float* __restrict__ out, int s)
{
  const int r = blockIdx.x, t = threadIdx.x;
  __shared__ float bval[256];
  __shared__ int   bidx[256];
  float best = -1e30f; int bi = 0;
  for (int j = t; j < VOCAB; j += 256) {
    float v = logits[(size_t)r * VOCAB + j];
    if (v > best) { best = v; bi = j; }
  }
  bval[t] = best; bidx[t] = bi; __syncthreads();
  for (int off = 128; off > 0; off >>= 1) {
    if (t < off) {
      if (bval[t + off] > bval[t] ||
          (bval[t + off] == bval[t] && bidx[t + off] < bidx[t])) {
        bval[t] = bval[t + off]; bidx[t] = bidx[t + off];
      }
    }
    __syncthreads();
  }
  const int tok = bidx[0];
  if (t == 0) out[(size_t)r * NSTEPS + s] = (float)tok;
  for (int j = t; j < DMODEL; j += 256)
    h[(size_t)r * DMODEL + j] = emb[(size_t)tok * DMODEL + j];
}

__global__ __launch_bounds__(256) void init_h_kernel(
    const float* __restrict__ src, bf16* __restrict__ h, int n)
{
  int i = blockIdx.x * 256 + threadIdx.x;
  if (i < n) h[i] = (bf16)src[i];
}

// ---------------------------------------------------------------------------
extern "C" void kernel_launch(void* const* d_in, const int* in_sizes, int n_in,
                              void* d_out, int out_size, void* d_ws, size_t ws_size,
                              hipStream_t stream) {
  const float* chunk = (const float*)d_in[0];
  const bf16*  wq    = (const bf16*)d_in[1];
  const bf16*  wk    = (const bf16*)d_in[2];
  const bf16*  wvw   = (const bf16*)d_in[3];
  const bf16*  wo    = (const bf16*)d_in[4];
  const bf16*  wg    = (const bf16*)d_in[5];
  const bf16*  wu    = (const bf16*)d_in[6];
  const bf16*  wd    = (const bf16*)d_in[7];
  const bf16*  ln1   = (const bf16*)d_in[8];
  const bf16*  ln2   = (const bf16*)d_in[9];
  const bf16*  wout  = (const bf16*)d_in[10];
  const bf16*  bout  = (const bf16*)d_in[11];
  const bf16*  emb   = (const bf16*)d_in[12];
  float* out = (float*)d_out;   // (B, T, NSTEPS) = 256 values

  char* ws = (char*)d_ws;
  size_t off = 0;
  auto alloc = [&](size_t bytes) -> void* {
    void* p = ws + off; off += (bytes + 255) & ~(size_t)255; return p;
  };
  bf16*  h   = (bf16*)alloc((size_t)ROWS * DMODEL * 2);
  bf16*  hn  = (bf16*)alloc((size_t)ROWS * DMODEL * 2);
  bf16*  qr  = (bf16*)alloc((size_t)ROWS * DMODEL * 2);
  bf16*  qv  = (bf16*)alloc((size_t)ROWS * DMODEL * 2);
  bf16*  kr  = (bf16*)alloc((size_t)ROWS * NKV * HDIM * 2);
  bf16*  kc  = (bf16*)alloc((size_t)ROWS * KVSTRIDE * 2);
  bf16*  vc  = (bf16*)alloc((size_t)ROWS * KVSTRIDE * 2);
  bf16*  ctx = (bf16*)alloc((size_t)ROWS * DMODEL * 2);
  bf16*  gu  = (bf16*)alloc((size_t)ROWS * DFF * 2);
  float* lg  = (float*)alloc((size_t)ROWS * VOCAB * 4);

  init_h_kernel<<<(ROWS * DMODEL + 255) / 256, 256, 0, stream>>>(chunk, h, ROWS * DMODEL);

  for (int s = 0; s < NSTEPS; ++s) {
    rmsnorm_kernel<<<ROWS, 256, 0, stream>>>(h, ln1, hn);
    gemm32<<<DMODEL / 128, 256, 0, stream>>>(hn, wq, DMODEL, DMODEL,
        qr, nullptr, nullptr, nullptr, DMODEL, 0);
    gemm32<<<(NKV * HDIM) / 128, 256, 0, stream>>>(hn, wk, DMODEL, NKV * HDIM,
        kr, nullptr, nullptr, nullptr, NKV * HDIM, 0);
    gemm32<<<(NKV * HDIM) / 128, 256, 0, stream>>>(hn, wvw, DMODEL, NKV * HDIM,
        vc + (size_t)s * NKV * HDIM, nullptr, nullptr, nullptr, KVSTRIDE, 0);
    rope_kernel<<<ROWS, 256, 0, stream>>>(qr, kr, qv, kc, s);
    attn_kernel<<<dim3(ROWS, NHEADS), 128, 0, stream>>>(qv, kc, vc, ctx, s);
    gemm32<<<DMODEL / 128, 256, 0, stream>>>(ctx, wo, DMODEL, DMODEL,
        h, nullptr, h, nullptr, DMODEL, 1);
    rmsnorm_kernel<<<ROWS, 256, 0, stream>>>(h, ln2, hn);
    gemm_gateup<<<DFF / 128, 256, 0, stream>>>(hn, wg, wu, gu);
    gemm32<<<DMODEL / 128, 256, 0, stream>>>(gu, wd, DFF, DMODEL,
        h, nullptr, h, nullptr, DMODEL, 1);
    gemm32<<<VOCAB / 128, 256, 0, stream>>>(h, wout, DMODEL, VOCAB,
        nullptr, lg, nullptr, bout, VOCAB, 2);
    argmax_embed_kernel<<<ROWS, 256, 0, stream>>>(lg, emb, h, out, s);
  }
}
